// Attention_45079976739000
// MI455X (gfx1250) — matrix operand order repaired and checked by reading, not compiled, not run on hardware
//
#include <hip/hip_runtime.h>
#include <hip/hip_bf16.h>

// Problem constants (match reference)
#define BB 16
#define TT 4096
#define DD 512
#define HH 64

typedef __attribute__((ext_vector_type(16))) _Float16 v16h;
typedef __attribute__((ext_vector_type(8)))  float    v8f;

#if defined(__gfx1250__) && __has_builtin(__builtin_amdgcn_global_load_async_to_lds_b128)
#define HAS_ASYNC_LDS 1
#else
#define HAS_ASYNC_LDS 0
#endif

// Builtin prototype (from hipcc diagnostic): params are pointers to
// int __attribute__((vector_size(16))) in AS1 (global) / AS3 (LDS).
typedef int vi4 __attribute__((vector_size(16)));
typedef __attribute__((address_space(1))) vi4* gvi4p;
typedef __attribute__((address_space(3))) vi4* lvi4p;

// Copy 16 bytes global -> LDS. Uses CDNA5 async-to-LDS when available.
static __device__ __forceinline__ void cp16_async(void* lds, const void* g) {
#if HAS_ASYNC_LDS
  __builtin_amdgcn_global_load_async_to_lds_b128(
      (gvi4p)(size_t)g,   // AS1: flat->global is numerically identity
      (lvi4p)lds,         // AS3: addrspacecast from generic
      0, 0);
#else
  *reinterpret_cast<uint4*>(lds) = *reinterpret_cast<const uint4*>(g);
#endif
}

static __device__ __forceinline__ void async_wait0() {
#if HAS_ASYNC_LDS && __has_builtin(__builtin_amdgcn_s_wait_asynccnt)
  __builtin_amdgcn_s_wait_asynccnt(0);
#elif HAS_ASYNC_LDS
  asm volatile("s_wait_asynccnt 0x0" ::: "memory");
#endif
}

static __device__ __forceinline__ v8f wmma16(v16h a, v16h b, v8f c) {
  // D = A(16x32 f16) x B(32x16 f16) + C(16x16 f32)
  return __builtin_amdgcn_wmma_f32_16x16x32_f16(false, a, false, b, (short)0, c,
                                                false, false);
}

static __device__ __forceinline__ v8f zero8() {
  v8f z;
#pragma unroll
  for (int i = 0; i < 8; ++i) z[i] = 0.0f;
  return z;
}

// Load 8 contiguous f16 (16B) into halves [base..base+7] of a v16h fragment.
static __device__ __forceinline__ void ld8h(const _Float16* p, v16h& dst, int base) {
  union { uint4 u; _Float16 h[8]; } cv;
  cv.u = *reinterpret_cast<const uint4*>(p);
#pragma unroll
  for (int i = 0; i < 8; ++i) dst[base + i] = cv.h[i];
}

// ---------------------------------------------------------------------------
// Kernel 0: one-shot weight transpose+convert.
// Wq/Wk/Wv [512,64] f32 (k-major)  ->  Wt [3][64][512] f16 (n-major rows).
// 192 KB total: L2-resident; makes every B-fragment load two b128 loads.
// ---------------------------------------------------------------------------
__global__ __launch_bounds__(256) void conv_w_kernel(
    const float* __restrict__ Wq, const float* __restrict__ Wk,
    const float* __restrict__ Wv, _Float16* __restrict__ Wt) {
  const int idx = blockIdx.x * 256 + threadIdx.x;   // 0 .. 3*64*512-1
  const int w   = idx / (HH * DD);
  const int rem = idx % (HH * DD);
  const int n   = rem / DD;
  const int kk  = rem % DD;
  const float* W = (w == 0) ? Wq : ((w == 1) ? Wk : Wv);
  Wt[idx] = (_Float16)W[(size_t)kk * HH + n];
}

// ---------------------------------------------------------------------------
// Kernel 1: fused QKV projection, fp32 x -> f16 q/k/vT.
// 4 waves/block; each wave owns 16 rows and all 12 N-tiles.
// B-fragments double-buffered so each WMMA only waits on loads issued one
// tile earlier (keeps 2 b128-pairs in flight instead of load->wait->wmma).
// ---------------------------------------------------------------------------
__global__ __launch_bounds__(128) void qkv_proj_kernel(
    const float* __restrict__ x, const _Float16* __restrict__ Wt,
    _Float16* __restrict__ q, _Float16* __restrict__ k,
    _Float16* __restrict__ vT) {
  const int lane  = threadIdx.x & 31;
  const int wave  = threadIdx.x >> 5;
  const int laneM = lane & 15;
  const int laneH = lane >> 4;
  const int rowBase = blockIdx.x * 64 + wave * 16;

  v8f acc[12];
#pragma unroll
  for (int i = 0; i < 12; ++i) acc[i] = zero8();

  const float* xr = x + (size_t)(rowBase + laneM) * DD;

  for (int kb = 0; kb < DD; kb += 32) {
    // A fragment (16x32 f16): lane = row laneM; halves: K = kb + laneH*16 + h
    const float* xp = xr + kb + laneH * 16;
    float4 f0 = *reinterpret_cast<const float4*>(xp);
    float4 f1 = *reinterpret_cast<const float4*>(xp + 4);
    float4 f2 = *reinterpret_cast<const float4*>(xp + 8);
    float4 f3 = *reinterpret_cast<const float4*>(xp + 12);
    v16h a;
    a[0]  = (_Float16)f0.x; a[1]  = (_Float16)f0.y; a[2]  = (_Float16)f0.z; a[3]  = (_Float16)f0.w;
    a[4]  = (_Float16)f1.x; a[5]  = (_Float16)f1.y; a[6]  = (_Float16)f1.z; a[7]  = (_Float16)f1.w;
    a[8]  = (_Float16)f2.x; a[9]  = (_Float16)f2.y; a[10] = (_Float16)f2.z; a[11] = (_Float16)f2.w;
    a[12] = (_Float16)f3.x; a[13] = (_Float16)f3.y; a[14] = (_Float16)f3.z; a[15] = (_Float16)f3.w;

    // Double-buffered B fragments from Wt[w][n][k]
    v16h bf0, bf1;
    {
      const _Float16* wp = Wt + ((size_t)laneM) * DD + kb + laneH * 16;  // t=0
      ld8h(wp, bf0, 0);
      ld8h(wp + 8, bf0, 8);
    }
#pragma unroll
    for (int t = 0; t < 12; ++t) {
      v16h& cur = (t & 1) ? bf1 : bf0;
      v16h& nxt = (t & 1) ? bf0 : bf1;
      if (t < 11) {
        const int t2  = t + 1;
        const int w2  = t2 >> 2;
        const int nb2 = (t2 & 3) * 16;
        const _Float16* wp =
            Wt + ((size_t)w2 * HH + nb2 + laneM) * DD + kb + laneH * 16;
        ld8h(wp, nxt, 0);
        ld8h(wp + 8, nxt, 8);
      }
      acc[t] = wmma16(a, cur, acc[t]);
    }
  }

  const int bIdx = rowBase / TT;
  const int t0   = rowBase % TT;
#pragma unroll
  for (int t = 0; t < 12; ++t) {
    const int w  = t >> 2;
    const int nb = (t & 3) * 16;
    if (w < 2) {
      _Float16* dst = (w == 0) ? q : k;
#pragma unroll
      for (int r = 0; r < 8; ++r) {
        const int row = rowBase + r + laneH * 8;   // C layout: M = r + 8*laneH
        dst[(size_t)row * HH + nb + laneM] = (_Float16)acc[t][r];
      }
    } else {
#pragma unroll
      for (int r = 0; r < 8; ++r) {
        const int tt2 = t0 + r + laneH * 8;
        vT[((size_t)bIdx * HH + nb + laneM) * TT + tt2] = (_Float16)acc[t][r];
      }
    }
  }
}

// ---------------------------------------------------------------------------
// Kernel 2: flash-attention forward (causal), f16 WMMA, f32 accumulation.
// Grid (T/64, B); 128 threads (4 waves); wave owns 16 query rows.
// Per 32-key block: K tile (32x64) and V tile (64x32, from vT) are staged
// into LDS cooperatively by the whole block via async-to-LDS (1x global
// traffic instead of 4x), then each wave does 4 WMMAs for S, online softmax,
// P transposed through a per-wave LDS tile, and 4 WMMAs for PV.
// ---------------------------------------------------------------------------
__global__ __launch_bounds__(128) void attn_kernel(
    const _Float16* __restrict__ q, const _Float16* __restrict__ k,
    const _Float16* __restrict__ vT, float* __restrict__ out) {
  constexpr int PSTR = 40;                        // padded row stride (halves)
  __shared__ alignas(16) _Float16 kbuf[32][HH];   // 4 KB: [key][hd]
  __shared__ alignas(16) _Float16 vbuf[HH][32];   // 4 KB: [hd][key]
  __shared__ alignas(16) _Float16 plds[4][16][PSTR];

  const int tid   = threadIdx.x;
  const int lane  = tid & 31;
  const int wave  = tid >> 5;
  const int laneM = lane & 15;
  const int laneH = lane >> 4;
  const int b     = blockIdx.y;
  const int qTile = blockIdx.x * 64;
  const int qBase = qTile + wave * 16;

  const _Float16* qb  = q  + (size_t)b * TT * HH;
  const _Float16* kb_ = k  + (size_t)b * TT * HH;
  const _Float16* vtb = vT + (size_t)b * HH * TT;

  // Q A-fragments for the two head-dim k-steps (hd 0..31 and 32..63)
  v16h aq0, aq1;
  {
    const _Float16* qr = qb + (size_t)(qBase + laneM) * HH;
    ld8h(qr +      laneH * 8, aq0, 0);
    ld8h(qr + 16 + laneH * 8, aq0, 8);
    ld8h(qr + 32 + laneH * 8, aq1, 0);
    ld8h(qr + 48 + laneH * 8, aq1, 8);
  }

  v8f o[4];
#pragma unroll
  for (int i = 0; i < 4; ++i) o[i] = zero8();
  float m[8], l[8];
#pragma unroll
  for (int r = 0; r < 8; ++r) { m[r] = -1e30f; l[r] = 0.0f; }

  const int kend      = qBase + 16;   // this wave's exclusive causal key limit
  const int kendBlock = qTile + 64;   // max over the block's waves

  for (int ks = 0; ks < kendBlock; ks += 32) {
    // ---- cooperative staging: K tile (contiguous 4KB) + V tile ----------
    {
      const char* kg = (const char*)(kb_ + (size_t)ks * HH);
      char* kl = (char*)&kbuf[0][0];
#pragma unroll
      for (int c = 0; c < 2; ++c) {
        const int off = (tid + c * 128) * 16;
        cp16_async(kl + off, kg + off);
      }
#pragma unroll
      for (int c = 0; c < 2; ++c) {
        const int u    = tid + c * 128;   // 16B unit of the 4KB V tile
        const int rowh = u >> 2;          // hd row 0..63
        const int seg  = u & 3;           // 16B segment within 64B row
        cp16_async(&vbuf[rowh][seg * 8],
                   vtb + (size_t)rowh * TT + ks + seg * 8);
      }
    }
    async_wait0();
    __syncthreads();

    if (ks < kend) {                       // wave-uniform compute guard
      // ---- S = Q K^T for two 16-key sub-tiles ---------------------------
      v8f s[2];
#pragma unroll
      for (int j = 0; j < 2; ++j) {
        const int kpos = ks + j * 16;
        if (kpos < kend) {                 // wave-uniform
          v16h bk0, bk1;                   // B frags: N=key, halves K=hd
          const _Float16* kr = &kbuf[j * 16 + laneM][laneH * 8];
          ld8h(kr,      bk0, 0); ld8h(kr + 16, bk0, 8);
          ld8h(kr + 32, bk1, 0); ld8h(kr + 48, bk1, 8);
          v8f sj = zero8();
          sj = wmma16(aq0, bk0, sj);
          sj = wmma16(aq1, bk1, sj);
          s[j] = sj;
        } else {
#pragma unroll
          for (int r = 0; r < 8; ++r) s[j][r] = -1e30f;
        }
      }

      // ---- scale, causal mask, online softmax ---------------------------
      const float scale = 0.125f;          // 1/sqrt(64)
      float p0[8], p1[8];
#pragma unroll
      for (int r = 0; r < 8; ++r) {
        const int row  = qBase + r + laneH * 8;
        const int key0 = ks + laneM;
        const int key1 = ks + 16 + laneM;
        float s0 = s[0][r] * scale; if (key0 > row) s0 = -1e30f;
        float s1 = s[1][r] * scale; if (key1 > row) s1 = -1e30f;
        float v = fmaxf(s0, s1);
#pragma unroll
        for (int off = 1; off < 16; off <<= 1) v = fmaxf(v, __shfl_xor(v, off, 16));
        const float mn    = fmaxf(m[r], v);
        const float alpha = __expf(m[r] - mn);
        m[r] = mn;
        const float e0 = __expf(s0 - mn);
        const float e1 = __expf(s1 - mn);
        float rs = e0 + e1;
#pragma unroll
        for (int off = 1; off < 16; off <<= 1) rs += __shfl_xor(rs, off, 16);
        l[r] = l[r] * alpha + rs;
        p0[r] = e0; p1[r] = e1;
#pragma unroll
        for (int nt = 0; nt < 4; ++nt) o[nt][r] *= alpha;
      }

      // ---- P: C-layout -> A-layout via per-wave LDS tile ----------------
#pragma unroll
      for (int r = 0; r < 8; ++r) {
        plds[wave][r + laneH * 8][laneM]      = (_Float16)p0[r];
        plds[wave][r + laneH * 8][16 + laneM] = (_Float16)p1[r];
      }
      v16h ap;
      {
        const _Float16* pr = &plds[wave][laneM][laneH * 8];
        ld8h(pr,      ap, 0);              // keys laneH*8 + 0..7
        ld8h(pr + 16, ap, 8);              // keys 16 + laneH*8 + 0..7
      }

      // ---- O += P x V  (vbuf rows contiguous in key position) -----------
#pragma unroll
      for (int nt = 0; nt < 4; ++nt) {
        v16h bv;                           // B frag: N=hd, halves K=key
        const _Float16* vr = &vbuf[nt * 16 + laneM][laneH * 8];
        ld8h(vr,      bv, 0);
        ld8h(vr + 16, bv, 8);
        o[nt] = wmma16(ap, bv, o[nt]);
      }
    }
    __syncthreads();                       // before next tile overwrite
  }

  // ---- epilogue: normalize and store fp32 -------------------------------
  float* ob = out + ((size_t)b * TT + qBase) * HH;
#pragma unroll
  for (int nt = 0; nt < 4; ++nt) {
#pragma unroll
    for (int r = 0; r < 8; ++r) {
      const int row = r + laneH * 8;
      ob[(size_t)row * HH + nt * 16 + laneM] = o[nt][r] / l[r];
    }
  }
}

extern "C" void kernel_launch(void* const* d_in, const int* in_sizes, int n_in,
                              void* d_out, int out_size, void* d_ws, size_t ws_size,
                              hipStream_t stream) {
  (void)in_sizes; (void)n_in; (void)out_size; (void)ws_size;
  const float* x  = (const float*)d_in[0];
  const float* Wq = (const float*)d_in[1];
  const float* Wk = (const float*)d_in[2];
  const float* Wv = (const float*)d_in[3];
  float* out = (float*)d_out;

  // Workspace: q [B*T,64] f16 | k [B*T,64] f16 | vT [B,64,T] f16 | Wt [3,64,512] f16
  _Float16* qws = (_Float16*)d_ws;
  _Float16* kws = qws + (size_t)BB * TT * HH;
  _Float16* vws = kws + (size_t)BB * TT * HH;
  _Float16* wts = vws + (size_t)BB * TT * HH;

  conv_w_kernel<<<(3 * HH * DD) / 256, 256, 0, stream>>>(Wq, Wk, Wv, wts);
  qkv_proj_kernel<<<(BB * TT) / 64, 128, 0, stream>>>(x, wts, qws, kws, vws);
  attn_kernel<<<dim3(TT / 64, BB), 128, 0, stream>>>(qws, kws, vws, out);
}
